// NativeLikeMambaBlock_30923764531650
// MI455X (gfx1250) — matrix operand order repaired and checked by reading, not compiled, not run on hardware
//
#include <hip/hip_runtime.h>
#include <cstddef>

// ---------------------------------------------------------------------------
// Model dims (fixed by the reference)
// ---------------------------------------------------------------------------
#define D_MODEL 768
#define D_INNER 1536
#define D_STATE 16
#define D_CONV  4
#define FFN_HID 1536
#define BATCH   2
#define SEQLEN  1024
#define MROWS   (BATCH * SEQLEN)   // 2048 token rows everywhere

// ---------------------------------------------------------------------------
// bf16 helpers (manual; no header struct dependencies)
// ---------------------------------------------------------------------------
__device__ __forceinline__ unsigned short f2bf(float f) {
  unsigned int u = __builtin_bit_cast(unsigned int, f);
  unsigned int r = u + 0x7FFFu + ((u >> 16) & 1u);   // RNE
  return (unsigned short)(r >> 16);
}
__device__ __forceinline__ float bf2f(unsigned short h) {
  unsigned int u = ((unsigned int)h) << 16;
  return __builtin_bit_cast(float, u);
}
__device__ __forceinline__ float sigmoidf_(float x) { return 1.0f / (1.0f + expf(-x)); }

// ---------------------------------------------------------------------------
// WMMA fragment / TDM descriptor types (CDNA5 / gfx1250, wave32)
// ---------------------------------------------------------------------------
typedef __bf16        bf16x16 __attribute__((ext_vector_type(16)));
typedef float         f32x8   __attribute__((ext_vector_type(8)));
typedef unsigned int  u32x4   __attribute__((ext_vector_type(4)));
typedef int           i32x4   __attribute__((ext_vector_type(4)));
typedef int           i32x8   __attribute__((ext_vector_type(8)));

union FragB16 {
  uint4   q[2];
  bf16x16 v;
};

// ---------------------------------------------------------------------------
// RMSNorm: one 256-thread block per token row; optional fp32 / bf16 outputs
// ---------------------------------------------------------------------------
__global__ void rmsnorm_kernel(const float* __restrict__ x,
                               const float* __restrict__ w,
                               float* __restrict__ outF,
                               unsigned short* __restrict__ outB, int d) {
  __shared__ float red[256];
  const int row = blockIdx.x;
  const float* xr = x + (size_t)row * d;
  float s = 0.0f;
  for (int c = threadIdx.x; c < d; c += 256) { float v = xr[c]; s += v * v; }
  red[threadIdx.x] = s;
  __syncthreads();
  for (int off = 128; off > 0; off >>= 1) {
    if (threadIdx.x < off) red[threadIdx.x] += red[threadIdx.x + off];
    __syncthreads();
  }
  const float scale = rsqrtf(red[0] / (float)d + 1e-5f);
  for (int c = threadIdx.x; c < d; c += 256) {
    const float v = xr[c] * scale * w[c];
    if (outF) outF[(size_t)row * d + c] = v;
    if (outB) outB[(size_t)row * d + c] = f2bf(v);
  }
}

// ---------------------------------------------------------------------------
// bf16-WMMA GEMM:  C[M,N] = act(A[M,K] * W[N,K]^T + bias) (+ residual)
//   A: bf16 row-major (pre-converted).  W: fp32 row-major, cvt while staging.
//   Block: 256 thr = 8 waves (4M x 2N); block tile 128x128; wave tile 32x64.
//   K-step 32 -> 8 x v_wmma_f32_16x16x32_bf16 per wave per step.
//   A tile (8KB bf16, 64B rows, stride 2K bytes) is DMA'd by the Tensor Data
//   Mover, software-pipelined one tile deep with double-buffered LDS:
//   TDM(step+1) streams in while WMMAs consume tile(step); in-order TDM
//   completion makes s_wait_tensorcnt(1) sufficient.
//   act: 0=none, 1=softplus(+1e-4), 2=exact GELU
// ---------------------------------------------------------------------------
__global__ void gemm_bf16_kernel(const unsigned short* __restrict__ Abf,
                                 const float* __restrict__ W,
                                 const float* __restrict__ bias,       // may be null
                                 const float* __restrict__ residual,   // may be null
                                 float* __restrict__ outF,             // may be null
                                 unsigned short* __restrict__ outB,    // may be null
                                 int M, int N, int K, int act) {
  // LDS byte layout (single static symbol, base offset 0):
  //   A0 @ 0, A1 @ 8192, W0 @ 16384, W1 @ 24576  (8KB tiles)
  __shared__ __align__(16) unsigned short smem[128 * 32 * 4];
  unsigned short* sW = smem + 2 * 128 * 32;

  const int blockN = blockIdx.x * 128;
  const int blockM = blockIdx.y * 128;

  const int tid  = threadIdx.x;
  const int wave = tid >> 5;       // 0..7
  const int lane = tid & 31;
  const int half = lane >> 4;      // lane half (ISA fragment layout)
  const int r    = lane & 15;
  const int wm   = wave >> 1;      // 0..3 -> 32-row M sub-tile
  const int wn   = wave & 1;       // 0..1 -> 64-col N sub-tile

  // W staging: thread -> (row 0..127, k-chunk of 16)
  const int srow = tid >> 1;
  const int sks  = (tid & 1) * 16;

  f32x8 acc[2][4];
#pragma unroll
  for (int a = 0; a < 2; ++a)
#pragma unroll
    for (int b = 0; b < 4; ++b)
#pragma unroll
      for (int j = 0; j < 8; ++j) acc[a][b][j] = 0.0f;

  const uint4* pA = reinterpret_cast<const uint4*>(smem);  // A buffers
  const uint4* pW = reinterpret_cast<const uint4*>(sW);    // W buffers

  const int nsteps = K / 32;

#if __has_builtin(__builtin_amdgcn_tensor_load_to_lds)
  // ---- TDM descriptor (all fields uniform -> SGPRs) ----
  const unsigned long long gbase =
      (unsigned long long)Abf + (unsigned long long)blockM * (unsigned long long)K * 2ull;
  i32x8 g1;
  g1[0] = (int)(1u << 16);                                    // data_size = 2B
  g1[1] = (int)(((unsigned)K & 0xFFFFu) << 16);               // tensor_dim0[15:0]
  g1[2] = (int)(((unsigned)K >> 16) | (((unsigned)M & 0xFFFFu) << 16)); // td0 hi | td1 lo
  g1[3] = (int)((((unsigned)M >> 16) & 0xFFFFu) | (32u << 16));         // td1 hi | tile_dim0=32
  g1[4] = (int)(128u);                                        // tile_dim1=128, tile_dim2=0
  g1[5] = (int)((unsigned)K);                                 // tensor_dim0_stride lo32
  g1[6] = 0;                                                  // stride0 hi | stride1 lo
  g1[7] = 0;                                                  // stride1 hi
  i32x4 gz4; gz4[0] = 0; gz4[1] = 0; gz4[2] = 0; gz4[3] = 0;
  i32x8 gz8;
#pragma unroll
  for (int j = 0; j < 8; ++j) gz8[j] = 0;

  auto tdm_issue = [&](int step) {
    const unsigned long long ga = gbase + (unsigned long long)step * 64ull; // 32 elem * 2B
    u32x4 g0;
    g0[0] = 1u;                                               // count=1
    g0[1] = (unsigned)((step & 1) ? 8192 : 0);                // lds_addr: A0/A1
    g0[2] = (unsigned)(ga & 0xFFFFFFFFull);
    g0[3] = (unsigned)((ga >> 32) & 0x1FFFFFFull) | (2u << 30);  // addr hi | type=2
    __builtin_amdgcn_tensor_load_to_lds(g0, g1, gz4, gz4, gz8, 0);
  };

  if (wave == 0) tdm_issue(0);   // prologue: tile 0 in flight
#endif

  for (int step = 0; step < nsteps; ++step) {
    const int kk = step * 32;
    const int cb = step & 1;     // current LDS buffer
    // ---- stage W tile [128 x 32] fp32 -> bf16 into sW[cb] ----
    {
      const float4* g = reinterpret_cast<const float4*>(
          W + (size_t)(blockN + srow) * K + kk + sks);
      float4 f0 = g[0], f1 = g[1], f2 = g[2], f3 = g[3];
      uint4 p0, p1;
      p0.x = (unsigned)f2bf(f0.x) | ((unsigned)f2bf(f0.y) << 16);
      p0.y = (unsigned)f2bf(f0.z) | ((unsigned)f2bf(f0.w) << 16);
      p0.z = (unsigned)f2bf(f1.x) | ((unsigned)f2bf(f1.y) << 16);
      p0.w = (unsigned)f2bf(f1.z) | ((unsigned)f2bf(f1.w) << 16);
      p1.x = (unsigned)f2bf(f2.x) | ((unsigned)f2bf(f2.y) << 16);
      p1.y = (unsigned)f2bf(f2.z) | ((unsigned)f2bf(f2.w) << 16);
      p1.z = (unsigned)f2bf(f3.x) | ((unsigned)f2bf(f3.y) << 16);
      p1.w = (unsigned)f2bf(f3.z) | ((unsigned)f2bf(f3.w) << 16);
      unsigned short* dst = &sW[cb * (128 * 32) + srow * 32 + sks];
      *reinterpret_cast<uint4*>(dst) = p0;
      *reinterpret_cast<uint4*>(dst + 8) = p1;
    }
#if __has_builtin(__builtin_amdgcn_tensor_load_to_lds)
    if (wave == 0) {
      if (step + 1 < nsteps) {
        tdm_issue(step + 1);                     // next tile streams during compute
        __builtin_amdgcn_s_wait_tensorcnt(1);    // in-order: tile(step) has landed
      } else {
        __builtin_amdgcn_s_wait_tensorcnt(0);
      }
    }
#else
    // fallback: cooperative copy of bf16 A tile into sA[cb]
    {
      const uint4* g = reinterpret_cast<const uint4*>(
          Abf + (size_t)(blockM + srow) * K + kk + sks);
      uint4 a0 = g[0], a1 = g[1];
      unsigned short* dst = &smem[cb * (128 * 32) + srow * 32 + sks];
      *reinterpret_cast<uint4*>(dst) = a0;
      *reinterpret_cast<uint4*>(dst + 8) = a1;
    }
#endif
    __syncthreads();   // tile(step) visible to all waves

    // ---- fragments (ISA 16-bit A 16x32 / B 32x16 layouts) ----
    const int abase = cb * 512;   // 8192B = 512 uint4
    FragB16 fa[2];
#pragma unroll
    for (int asub = 0; asub < 2; ++asub) {
      const int rowA = wm * 32 + asub * 16 + r;
      fa[asub].q[0] = pA[abase + rowA * 4 + half];       // K = 8h..8h+7
      fa[asub].q[1] = pA[abase + rowA * 4 + 2 + half];   // K = 16+8h..23+8h
    }
#pragma unroll
    for (int bsub = 0; bsub < 4; ++bsub) {
      FragB16 fb;
      const int nrow = wn * 64 + bsub * 16 + r;
      fb.q[0] = pW[abase + nrow * 4 + half];             // K = 8h..8h+7
      fb.q[1] = pW[abase + nrow * 4 + 2 + half];         // K = 16+8h..23+8h
      acc[0][bsub] = __builtin_amdgcn_wmma_f32_16x16x32_bf16(
          false, fa[0].v, false, fb.v, (short)0, acc[0][bsub], false, false);
      acc[1][bsub] = __builtin_amdgcn_wmma_f32_16x16x32_bf16(
          false, fa[1].v, false, fb.v, (short)0, acc[1][bsub], false, false);
    }
    __syncthreads();   // reads done before buffer is re-filled
  }

  // ---- epilogue ----
#pragma unroll
  for (int asub = 0; asub < 2; ++asub) {
#pragma unroll
    for (int bsub = 0; bsub < 4; ++bsub) {
      const int n  = blockN + wn * 64 + bsub * 16 + r;
      const float bv = bias ? bias[n] : 0.0f;
#pragma unroll
      for (int j = 0; j < 8; ++j) {
        const int m = blockM + wm * 32 + asub * 16 + j + 8 * half;
        float v = acc[asub][bsub][j] + bv;
        if (act == 1) {                               // softplus + 1e-4
          v = fmaxf(v, 0.0f) + log1pf(expf(-fabsf(v))) + 1e-4f;
        } else if (act == 2) {                        // exact GELU
          v = 0.5f * v * (1.0f + erff(v * 0.70710678118654752f));
        }
        const size_t idx = (size_t)m * N + n;
        if (residual) v += residual[idx];
        if (outF) outF[idx] = v;
        if (outB) outB[idx] = f2bf(v);
      }
    }
  }
}

// ---------------------------------------------------------------------------
// Depthwise causal conv1d (width 4) + SiLU -> bf16
// ---------------------------------------------------------------------------
__global__ void conv_silu_kernel(const float* __restrict__ xz,   // [M, 2*D_INNER]
                                 const float* __restrict__ cw,   // [D_INNER,1,4]
                                 const float* __restrict__ cb,   // [D_INNER]
                                 unsigned short* __restrict__ u) { // [M, D_INNER] bf16
  const int idx = blockIdx.x * blockDim.x + threadIdx.x;
  if (idx >= MROWS * D_INNER) return;
  const int m = idx / D_INNER;
  const int c = idx - m * D_INNER;
  const int t = m & (SEQLEN - 1);
  float s = cb[c];
#pragma unroll
  for (int j = 0; j < D_CONV; ++j) {
    const int tt = t - (D_CONV - 1) + j;
    if (tt >= 0)
      s += xz[(size_t)(m - (D_CONV - 1) + j) * (2 * D_INNER) + c] * cw[c * D_CONV + j];
  }
  const float v = s * sigmoidf_(s);
  u[idx] = f2bf(v);
}

// ---------------------------------------------------------------------------
// Selective scan: one 16-lane group per (batch, channel); state in registers.
// ---------------------------------------------------------------------------
__global__ void scan_kernel(const float* __restrict__ dt,   // [M, D_INNER]
                            const float* __restrict__ u2,   // [M, D_INNER]
                            const unsigned short* __restrict__ Bt,  // [M, D_INNER*16] bf16
                            const unsigned short* __restrict__ Ct,
                            const float* __restrict__ A_log, // [D_INNER,16]
                            const float* __restrict__ Dp,    // [D_INNER]
                            float* __restrict__ y) {         // [M, D_INNER]
  const int s = threadIdx.x & 15;
  const int i = blockIdx.x * 16 + (threadIdx.x >> 4);
  const int b = blockIdx.y;
  const float a  = -expf(A_log[i * D_STATE + s]);
  const float dv = Dp[i];
  float state = 0.0f;
  for (int t = 0; t < SEQLEN; ++t) {
    const int m = b * SEQLEN + t;
    const float dtv = dt[(size_t)m * D_INNER + i];
    const float uv  = u2[(size_t)m * D_INNER + i];
    const float Bv  = bf2f(Bt[(size_t)m * (D_INNER * D_STATE) + i * D_STATE + s]);
    const float Cv  = bf2f(Ct[(size_t)m * (D_INNER * D_STATE) + i * D_STATE + s]);
    state = expf(dtv * a) * state + Bv * uv;
    float p = state * Cv;
    p += __shfl_xor(p, 8, 16);
    p += __shfl_xor(p, 4, 16);
    p += __shfl_xor(p, 2, 16);
    p += __shfl_xor(p, 1, 16);
    if (s == 0) y[(size_t)m * D_INNER + i] = p + dv * uv;
  }
}

// ---------------------------------------------------------------------------
// Gate: yg = y * sigmoid(z) -> bf16   (z = second half of xz)
// ---------------------------------------------------------------------------
__global__ void gate_kernel(const float* __restrict__ y,
                            const float* __restrict__ xz,
                            unsigned short* __restrict__ yg) {
  const int idx = blockIdx.x * blockDim.x + threadIdx.x;
  if (idx >= MROWS * D_INNER) return;
  const int m = idx / D_INNER;
  const int c = idx - m * D_INNER;
  const float z = xz[(size_t)m * (2 * D_INNER) + D_INNER + c];
  yg[idx] = f2bf(y[idx] * sigmoidf_(z));
}

// ---------------------------------------------------------------------------
// Launcher
// ---------------------------------------------------------------------------
extern "C" void kernel_launch(void* const* d_in, const int* in_sizes, int n_in,
                              void* d_out, int out_size, void* d_ws, size_t ws_size,
                              hipStream_t stream) {
  (void)in_sizes; (void)n_in; (void)out_size; (void)ws_size;
  const float* x        = (const float*)d_in[0];
  const float* norm_w   = (const float*)d_in[1];
  const float* in_w     = (const float*)d_in[2];
  const float* in_b     = (const float*)d_in[3];
  const float* conv_w   = (const float*)d_in[4];
  const float* conv_b   = (const float*)d_in[5];
  const float* xproj_w  = (const float*)d_in[6];
  const float* dtproj_w = (const float*)d_in[7];
  const float* dtproj_b = (const float*)d_in[8];
  const float* bproj_w  = (const float*)d_in[9];
  const float* cproj_w  = (const float*)d_in[10];
  const float* A_log    = (const float*)d_in[11];
  const float* D_param  = (const float*)d_in[12];
  const float* out_w    = (const float*)d_in[13];
  const float* out_b    = (const float*)d_in[14];
  const float* ffn_nw   = (const float*)d_in[15];
  const float* ffn_w1   = (const float*)d_in[16];
  const float* ffn_b1   = (const float*)d_in[17];
  const float* ffn_w2   = (const float*)d_in[18];
  const float* ffn_b2   = (const float*)d_in[19];
  float* out = (float*)d_out;

  // ---- workspace carve-out ----
  char* ws = (char*)d_ws;
  size_t cur = 0;
  auto alloc = [&](size_t bytes) { char* p = ws + cur; cur += (bytes + 255) & ~(size_t)255; return p; };
  unsigned short* h_bf   = (unsigned short*)alloc((size_t)MROWS * D_MODEL * 2);
  float*          xz     = (float*)alloc((size_t)MROWS * 2 * D_INNER * 4);
  unsigned short* ucv_bf = (unsigned short*)alloc((size_t)MROWS * D_INNER * 2);
  float*          u2f    = (float*)alloc((size_t)MROWS * D_INNER * 4);
  unsigned short* u2b    = (unsigned short*)alloc((size_t)MROWS * D_INNER * 2);
  float*          dtf    = (float*)alloc((size_t)MROWS * D_INNER * 4);
  float*          yb     = (float*)alloc((size_t)MROWS * D_INNER * 4);
  unsigned short* yg_bf  = (unsigned short*)alloc((size_t)MROWS * D_INNER * 2);
  float*          xmid   = (float*)alloc((size_t)MROWS * D_MODEL * 4);
  unsigned short* h2_bf  = (unsigned short*)alloc((size_t)MROWS * D_MODEL * 2);
  unsigned short* ff_bf  = (unsigned short*)alloc((size_t)MROWS * FFN_HID * 2);
  unsigned short* Btb    = (unsigned short*)alloc((size_t)MROWS * D_INNER * D_STATE * 2);
  unsigned short* Ctb    = (unsigned short*)alloc((size_t)MROWS * D_INNER * D_STATE * 2);

  const dim3 blk256(256);
  auto gemm = [&](const unsigned short* A, const float* W, const float* bias,
                  const float* res, float* oF, unsigned short* oB,
                  int M, int N, int K, int act) {
    dim3 grid(N / 128, M / 128);
    gemm_bf16_kernel<<<grid, blk256, 0, stream>>>(A, W, bias, res, oF, oB, M, N, K, act);
  };

  // 1) h = rmsnorm(x)  (bf16 for GEMM A)
  rmsnorm_kernel<<<MROWS, blk256, 0, stream>>>(x, norm_w, nullptr, h_bf, D_MODEL);
  // 2) xz = h @ in_proj^T + b
  gemm(h_bf, in_w, in_b, nullptr, xz, nullptr, MROWS, 2 * D_INNER, D_MODEL, 0);
  // 3) u = silu(depthwise causal conv(x-branch)) -> bf16
  {
    int n = MROWS * D_INNER;
    conv_silu_kernel<<<(n + 255) / 256, blk256, 0, stream>>>(xz, conv_w, conv_b, ucv_bf);
  }
  // 4) u2 = u @ x_proj^T  (fp32 for scan + bf16 for downstream GEMMs)
  gemm(ucv_bf, xproj_w, nullptr, nullptr, u2f, u2b, MROWS, D_INNER, D_INNER, 0);
  // 5) dt = softplus(u2 @ dt_proj^T + b) + 1e-4
  gemm(u2b, dtproj_w, dtproj_b, nullptr, dtf, nullptr, MROWS, D_INNER, D_INNER, 1);
  // 6,7) Bt / Ct (bf16, consumed by the scan)
  gemm(u2b, bproj_w, nullptr, nullptr, nullptr, Btb, MROWS, D_INNER * D_STATE, D_INNER, 0);
  gemm(u2b, cproj_w, nullptr, nullptr, nullptr, Ctb, MROWS, D_INNER * D_STATE, D_INNER, 0);
  // 8) selective scan
  {
    dim3 grid(D_INNER / 16, BATCH);
    scan_kernel<<<grid, blk256, 0, stream>>>(dtf, u2f, Btb, Ctb, A_log, D_param, yb);
  }
  // 9) gate with sigmoid(z) -> bf16
  {
    int n = MROWS * D_INNER;
    gate_kernel<<<(n + 255) / 256, blk256, 0, stream>>>(yb, xz, yg_bf);
  }
  // 10) x_mid = x + yg @ out_proj^T + b
  gemm(yg_bf, out_w, out_b, x, xmid, nullptr, MROWS, D_MODEL, D_INNER, 0);
  // 11) h2 = rmsnorm(x_mid) -> bf16
  rmsnorm_kernel<<<MROWS, blk256, 0, stream>>>(xmid, ffn_nw, nullptr, h2_bf, D_MODEL);
  // 12) ffhid = gelu(h2 @ w1^T + b1) -> bf16
  gemm(h2_bf, ffn_w1, ffn_b1, nullptr, nullptr, ff_bf, MROWS, FFN_HID, D_MODEL, 2);
  // 13) out = x_mid + ffhid @ w2^T + b2
  gemm(ff_bf, ffn_w2, ffn_b2, xmid, out, nullptr, MROWS, D_MODEL, FFN_HID, 0);
}
